// MultiscaleAttention_20804821581901
// MI455X (gfx1250) — matrix operand order repaired and checked by reading, not compiled, not run on hardware
//
#include <hip/hip_runtime.h>

typedef __bf16 bf16;
typedef __attribute__((ext_vector_type(16))) __bf16 v16bf;
typedef __attribute__((ext_vector_type(8)))  __bf16 v8bf;
typedef __attribute__((ext_vector_type(8)))  float   v8f;
typedef __attribute__((ext_vector_type(4)))  unsigned int u32x4;
typedef __attribute__((ext_vector_type(8)))  unsigned int u32x8;

#define D_MODEL 1024
#define NHEAD   16
#define HDIM    64

union FragBF { v16bf v; v8bf h[2]; __bf16 e[16]; };
union FragF  { v8f   v; float f[8]; };

// ---------------- elementwise f32 -> bf16 ----------------
__global__ __launch_bounds__(256)
void k_cvt_bf16(const float* __restrict__ in, bf16* __restrict__ out, int n) {
  int i = blockIdx.x * 256 + threadIdx.x;
  if (i < n) out[i] = (bf16)in[i];
}

// ---------------- avg-pool rows (factor pool) then f32 -> bf16 ----------------
__global__ __launch_bounds__(256)
void k_pool_cvt(const float* __restrict__ vbase, bf16* __restrict__ out,
                int rowsOut, int pool) {
  int i = blockIdx.x * 256 + threadIdx.x;
  if (i >= rowsOut * D_MODEL) return;
  int col = i & (D_MODEL - 1);
  int row = i >> 10;
  float s = 0.f;
  for (int p = 0; p < pool; ++p)
    s += vbase[(size_t)(row * pool + p) * D_MODEL + col];
  out[i] = (bf16)(s * (1.0f / (float)pool));
}

// ---------------- RoPE (fp32) -> bf16 Q/K;  Q pre-scaled by 1/sqrt(64) ----------------
__global__ __launch_bounds__(256)
void k_rope_cvt(const float* __restrict__ q, const float* __restrict__ k,
                bf16* __restrict__ qb, bf16* __restrict__ kb,
                int nrows, float posScale) {
  int i = blockIdx.x * 256 + threadIdx.x;
  if (i >= nrows * NHEAD * (HDIM / 2)) return;
  int j = i & 31;
  int h = (i >> 5) & (NHEAD - 1);
  int n = i >> 9;
  // inv_freq = 10000^(-2j/64) = exp(-j * ln(10000)/32)
  float inv = __expf(-(float)j * 0.28782313662425575f);
  float ang = (float)n * posScale * inv;
  float s, c;
  __sincosf(ang, &s, &c);
  size_t base = (size_t)n * D_MODEL + (size_t)h * HDIM + j;
  float q1 = q[base], q2 = q[base + 32];
  qb[base]      = (bf16)((q1 * c - q2 * s) * 0.125f);
  qb[base + 32] = (bf16)((q1 * s + q2 * c) * 0.125f);
  float k1 = k[base], k2 = k[base + 32];
  kb[base]      = (bf16)(k1 * c - k2 * s);
  kb[base + 32] = (bf16)(k1 * s + k2 * c);
}

// A 16x32 layout: lanes<16 hold K 0..7 & 16..23; lanes>=16 hold K 8..15 & 24..31
#define LOAD_FA(f, base, k)                                            \
  { (f).h[0] = *(const v8bf*)((base) + (k) + half * 8);                \
    (f).h[1] = *(const v8bf*)((base) + (k) + 16 + half * 8); }
// B 32x16 layout: lanes<16 hold K 0..7 & 16..23; lanes>=16 hold K 8..15 & 24..31
#define LOAD_FB(f, base, k)                                            \
  { (f).h[0] = *(const v8bf*)((base) + (k) + half * 8);                \
    (f).h[1] = *(const v8bf*)((base) + (k) + 16 + half * 8); }

// ---------------- GEMM: C[M,Nc] = A[M,Dk](bf16) * W[Nc,Dk]^T(bf16) + bias, fp32 out ----
// 4 waves / block; each wave owns a 32x32 tile (2x2 WMMAs); K unrolled x2 with
// ping-pong fragment buffers (no inter-stage register copies). Dk % 64 == 0.
__global__ __launch_bounds__(128)
void k_gemm_bf16(const bf16* __restrict__ A, const bf16* __restrict__ W,
                 const float* __restrict__ bias, float* __restrict__ C,
                 int M, int Dk, int Nc) {
  int lane = threadIdx.x & 31;
  int wave = threadIdx.x >> 5;
  int ntN  = Nc >> 5;
  int tile = blockIdx.x * 4 + wave;
  if (tile >= (M >> 5) * ntN) return;
  int m0 = (tile / ntN) << 5;
  int n0 = (tile % ntN) << 5;
  int half = lane >> 4, lr = lane & 15;

  const bf16* Ar0 = A + (size_t)(m0 + lr) * Dk;        // A-frag rows (lane%16)
  const bf16* Ar1 = A + (size_t)(m0 + 16 + lr) * Dk;
  const bf16* Wr0 = W + (size_t)(n0 + lr) * Dk;        // B-frag cols (B = W^T)
  const bf16* Wr1 = W + (size_t)(n0 + 16 + lr) * Dk;

  FragF acc00, acc01, acc10, acc11;
  acc00.v = (v8f){}; acc01.v = (v8f){}; acc10.v = (v8f){}; acc11.v = (v8f){};

  FragBF a0[2], a1[2], b0[2], b1[2];
  LOAD_FA(a0[0], Ar0, 0); LOAD_FA(a1[0], Ar1, 0);
  LOAD_FB(b0[0], Wr0, 0); LOAD_FB(b1[0], Wr1, 0);

  for (int k0 = 0; k0 < Dk; k0 += 64) {
    // stage 1: load frags[1] @ k0+32 (always in range: Dk % 64 == 0)
    LOAD_FA(a0[1], Ar0, k0 + 32); LOAD_FA(a1[1], Ar1, k0 + 32);
    LOAD_FB(b0[1], Wr0, k0 + 32); LOAD_FB(b1[1], Wr1, k0 + 32);
    __builtin_prefetch((const void*)(Ar0 + k0 + 128), 0, 1);
    __builtin_prefetch((const void*)(Wr0 + k0 + 128), 0, 1);
    acc00.v = __builtin_amdgcn_wmma_f32_16x16x32_bf16(false, a0[0].v, false, b0[0].v, (short)0, acc00.v, false, false);
    acc01.v = __builtin_amdgcn_wmma_f32_16x16x32_bf16(false, a0[0].v, false, b1[0].v, (short)0, acc01.v, false, false);
    acc10.v = __builtin_amdgcn_wmma_f32_16x16x32_bf16(false, a1[0].v, false, b0[0].v, (short)0, acc10.v, false, false);
    acc11.v = __builtin_amdgcn_wmma_f32_16x16x32_bf16(false, a1[0].v, false, b1[0].v, (short)0, acc11.v, false, false);
    // stage 2: load frags[0] @ k0+64 (guarded tail), consume frags[1]
    if (k0 + 64 < Dk) {
      LOAD_FA(a0[0], Ar0, k0 + 64); LOAD_FA(a1[0], Ar1, k0 + 64);
      LOAD_FB(b0[0], Wr0, k0 + 64); LOAD_FB(b1[0], Wr1, k0 + 64);
    }
    acc00.v = __builtin_amdgcn_wmma_f32_16x16x32_bf16(false, a0[1].v, false, b0[1].v, (short)0, acc00.v, false, false);
    acc01.v = __builtin_amdgcn_wmma_f32_16x16x32_bf16(false, a0[1].v, false, b1[1].v, (short)0, acc01.v, false, false);
    acc10.v = __builtin_amdgcn_wmma_f32_16x16x32_bf16(false, a1[1].v, false, b0[1].v, (short)0, acc10.v, false, false);
    acc11.v = __builtin_amdgcn_wmma_f32_16x16x32_bf16(false, a1[1].v, false, b1[1].v, (short)0, acc11.v, false, false);
  }

  float bc0 = bias[n0 + lr];
  float bc1 = bias[n0 + 16 + lr];
  for (int r = 0; r < 8; ++r) {
    int row0 = m0 + r + half * 8;       // C layout: VGPR r -> rows r / r+8
    int row1 = row0 + 16;
    C[(size_t)row0 * Nc + n0 + lr]      = acc00.f[r] + bc0;
    C[(size_t)row0 * Nc + n0 + 16 + lr] = acc01.f[r] + bc1;
    C[(size_t)row1 * Nc + n0 + lr]      = acc10.f[r] + bc0;
    C[(size_t)row1 * Nc + n0 + 16 + lr] = acc11.f[r] + bc1;
  }
}

// ---------------- Flash attention: one wave per (head, 16-query tile) ----------------
// V tiles are DMA'd to LDS by the Tensor Data Mover (tensor_load_to_lds), overlapped
// with the Q*K^T WMMAs and the online softmax; synchronized with s_wait_tensorcnt.
__global__ __launch_bounds__(32)
void k_flash_attn(const bf16* __restrict__ Q, const bf16* __restrict__ K,
                  const bf16* __restrict__ V, float* __restrict__ O, int N) {
  __shared__ float sP[16 * 32];   // P tile staging (C-layout -> A-layout transpose)
  __shared__ bf16  sV[32 * 64];   // V tile (32 keys x 64 dims), filled by TDM

  int qt = blockIdx.x, h = blockIdx.y;
  int lane = threadIdx.x;
  int half = lane >> 4, lr = lane & 15;
  const size_t hoff = (size_t)h * HDIM;

  // Q A-fragments (d = 0..31, 32..63)
  const bf16* qrow = Q + (size_t)(qt * 16 + lr) * D_MODEL + hoff;
  FragBF qa0, qa1;
  LOAD_FA(qa0, qrow, 0);
  LOAD_FA(qa1, qrow, 32);

  float m_i[8], l_i[8];
  FragF o[4];
  for (int r = 0; r < 8; ++r) { m_i[r] = -1e30f; l_i[r] = 0.f; }
  for (int t = 0; t < 4; ++t) o[t].v = (v8f){};

  const unsigned ldsbase = (unsigned)(uintptr_t)(void*)&sV[0]; // low 32 bits = LDS offset
  int nkeys = (qt + 1) * 16;                                    // causal limit

  for (int kb = 0; kb < nkeys; kb += 32) {
    // ---- TDM: DMA V tile [min(32,N-kb) x 64] (bf16, row stride 1024) into LDS ----
    // D# group0: count=1 | lds_addr | global_addr[56:0] | type=2
    // D# group1: data_size=2B; tensor_dim0=64; tensor_dim1=N-kb (tail rows zero-filled);
    //            tile_dim0=64, tile_dim1=32; tensor_dim0_stride=1024
    {
      unsigned long long ga = (unsigned long long)(uintptr_t)(V + (size_t)kb * D_MODEL + hoff);
      unsigned rows = (unsigned)(N - kb);
      u32x4 g0 = { 1u, ldsbase, (unsigned)(ga & 0xFFFFFFFFu),
                   (unsigned)((ga >> 32) & 0x01FFFFFFu) | (2u << 30) };
      u32x8 g1 = { 0x00010000u,        // data_size = 2 bytes
                   64u << 16,          // tensor_dim0 = 64
                   rows << 16,         // tensor_dim1 = remaining rows
                   64u << 16,          // tile_dim0 = 64
                   32u,                // tile_dim1 = 32 (tile_dim2 = 0)
                   1024u,              // tensor_dim0_stride = 1024 elements
                   0u, 0u };
      asm volatile("tensor_load_to_lds %0, %1" :: "s"(g0), "s"(g1) : "memory");
    }

    // ---- S = Q K^T for two 16-key subtiles (overlaps the TDM) ----
    FragF sc[2];
    for (int t2 = 0; t2 < 2; ++t2) {
      int krow = kb + t2 * 16 + lr; if (krow > N - 1) krow = N - 1;
      const bf16* kr = K + (size_t)krow * D_MODEL + hoff;
      FragBF kb0, kb1;
      LOAD_FB(kb0, kr, 0);
      LOAD_FB(kb1, kr, 32);
      v8f z = {};
      z = __builtin_amdgcn_wmma_f32_16x16x32_bf16(false, qa0.v, false, kb0.v, (short)0, z, false, false);
      z = __builtin_amdgcn_wmma_f32_16x16x32_bf16(false, qa1.v, false, kb1.v, (short)0, z, false, false);
      sc[t2].v = z;
    }
    // ---- causal mask + online softmax (row reductions across 16-lane halves) ----
    float mnew[8];
    for (int r = 0; r < 8; ++r) {
      int qg = qt * 16 + r + half * 8;
      float s0 = sc[0].f[r];
      float s1 = sc[1].f[r];
      if (kb + lr      > qg || kb + lr      >= N) s0 = -1e30f;
      if (kb + 16 + lr > qg || kb + 16 + lr >= N) s1 = -1e30f;
      sc[0].f[r] = s0; sc[1].f[r] = s1;
      float mx = fmaxf(s0, s1);
      for (int d = 1; d < 16; d <<= 1) mx = fmaxf(mx, __shfl_xor(mx, d, 32));
      mnew[r] = fmaxf(m_i[r], mx);
    }
    for (int r = 0; r < 8; ++r) {
      float p0 = __expf(sc[0].f[r] - mnew[r]);
      float p1 = __expf(sc[1].f[r] - mnew[r]);
      sc[0].f[r] = p0; sc[1].f[r] = p1;
      float rs = p0 + p1;
      for (int d = 1; d < 16; d <<= 1) rs += __shfl_xor(rs, d, 32);
      float corr = __expf(m_i[r] - mnew[r]);
      l_i[r] = l_i[r] * corr + rs;
      m_i[r] = mnew[r];
      for (int t = 0; t < 4; ++t) o[t].f[r] *= corr;
    }
    // ---- P: C-layout -> LDS -> A-layout (bf16) ----
    for (int r = 0; r < 8; ++r) {
      int prow = r + half * 8;
      sP[prow * 32 + lr]      = sc[0].f[r];
      sP[prow * 32 + 16 + lr] = sc[1].f[r];
    }
    __syncthreads();
    FragBF pa;
    {
      const float* pr = &sP[lr * 32];
      for (int jj = 0; jj < 8; ++jj) {
        pa.e[jj]     = (bf16)pr[half * 16 + jj];
        pa.e[8 + jj] = (bf16)pr[half * 16 + 8 + jj];
      }
    }
    // ---- wait for TDM, then O += P * V (V gathered from LDS in B-fragment layout) ----
    __builtin_amdgcn_s_wait_tensorcnt(0);
    for (int t = 0; t < 4; ++t) {
      FragBF vb;
      for (int jj = 0; jj < 8; ++jj) {
        int kr0 = 2 * jj + half * 16;
        vb.e[2 * jj]     = sV[kr0 * 64 + t * 16 + lr];
        vb.e[2 * jj + 1] = sV[(kr0 + 1) * 64 + t * 16 + lr];
      }
      o[t].v = __builtin_amdgcn_wmma_f32_16x16x32_bf16(
          false, pa.v, false, vb.v, (short)0, o[t].v, false, false);
    }
    __syncthreads();
  }
  // ---- epilogue: normalize and store fp32 context ----
  for (int r = 0; r < 8; ++r) {
    float inv_l = 1.f / l_i[r];
    int row = qt * 16 + r + half * 8;
    float* orow = O + (size_t)row * D_MODEL + hoff;
    for (int t = 0; t < 4; ++t) orow[t * 16 + lr] = o[t].f[r] * inv_l;
  }
}

// ---------------- host side ----------------
static inline int cdiv(int a, int b) { return (a + b - 1) / b; }

extern "C" void kernel_launch(void* const* d_in, const int* in_sizes, int n_in,
                              void* d_out, int out_size, void* d_ws, size_t ws_size,
                              hipStream_t stream) {
  (void)in_sizes; (void)n_in; (void)out_size; (void)ws_size;
  const float* x[3] = {(const float*)d_in[0], (const float*)d_in[1], (const float*)d_in[2]};
  const float* Wq = (const float*)d_in[3];
  const float* bq = (const float*)d_in[4];
  const float* Wk = (const float*)d_in[5];
  const float* bk = (const float*)d_in[6];
  const float* Wv = (const float*)d_in[7];
  const float* bv = (const float*)d_in[8];
  const float* Wo = (const float*)d_in[9];
  const float* bo = (const float*)d_in[10];
  float* out = (float*)d_out;

  const int Nsc[3] = {2048, 1024, 512};
  const int DD = D_MODEL * D_MODEL;          // 1M

  // ---- workspace carve-up (256B aligned) ----
  char* w = (char*)d_ws;
  auto carve = [&](size_t bytes) -> char* {
    char* p = w; w += (bytes + 255) & ~(size_t)255; return p;
  };
  bf16* xb[3];
  xb[0] = (bf16*)carve((size_t)2048 * D_MODEL * 2);
  xb[1] = (bf16*)carve((size_t)1024 * D_MODEL * 2);
  xb[2] = (bf16*)carve((size_t)512  * D_MODEL * 2);
  bf16* Wq_b = (bf16*)carve((size_t)3 * DD * 2);
  bf16* Wk_b = (bf16*)carve((size_t)3 * DD * 2);
  bf16* Wv_b = (bf16*)carve((size_t)DD * 2);
  bf16* Wo_b = (bf16*)carve((size_t)DD * 2);
  float* vbase = (float*)carve((size_t)2048 * D_MODEL * 4);
  bf16* vb[3];
  vb[0] = (bf16*)carve((size_t)2048 * D_MODEL * 2);
  vb[1] = (bf16*)carve((size_t)1024 * D_MODEL * 2);
  vb[2] = (bf16*)carve((size_t)512  * D_MODEL * 2);
  float* qf   = (float*)carve((size_t)2048 * D_MODEL * 4);
  float* kf   = (float*)carve((size_t)2048 * D_MODEL * 4);
  bf16* qb    = (bf16*)carve((size_t)2048 * D_MODEL * 2);
  bf16* kb2   = (bf16*)carve((size_t)2048 * D_MODEL * 2);
  float* ctx  = (float*)carve((size_t)2048 * D_MODEL * 4);
  bf16* ctxb  = (bf16*)carve((size_t)2048 * D_MODEL * 2);

  // ---- 1) fp32 -> bf16 conversions ----
  for (int i = 0; i < 3; ++i) {
    int n = Nsc[i] * D_MODEL;
    k_cvt_bf16<<<cdiv(n, 256), 256, 0, stream>>>(x[i], xb[i], n);
  }
  k_cvt_bf16<<<cdiv(3 * DD, 256), 256, 0, stream>>>(Wq, Wq_b, 3 * DD);
  k_cvt_bf16<<<cdiv(3 * DD, 256), 256, 0, stream>>>(Wk, Wk_b, 3 * DD);
  k_cvt_bf16<<<cdiv(DD, 256), 256, 0, stream>>>(Wv, Wv_b, DD);
  k_cvt_bf16<<<cdiv(DD, 256), 256, 0, stream>>>(Wo, Wo_b, DD);

  // ---- 2) V projection at scale 0 ----
  {
    int tiles = (2048 / 32) * (D_MODEL / 32);
    k_gemm_bf16<<<cdiv(tiles, 4), 128, 0, stream>>>(xb[0], Wv_b, bv, vbase,
                                                    2048, D_MODEL, D_MODEL);
  }
  // ---- 3) per-scale V (avg-pool + bf16) ----
  for (int i = 0; i < 3; ++i) {
    int n = Nsc[i] * D_MODEL;
    k_pool_cvt<<<cdiv(n, 256), 256, 0, stream>>>(vbase, vb[i], Nsc[i], 1 << i);
  }

  // ---- 4..7) per-scale pipeline ----
  size_t out_off = 0;
  for (int i = 0; i < 3; ++i) {
    int Ni = Nsc[i];
    int tiles = (Ni / 32) * (D_MODEL / 32);
    // Q/K projections
    k_gemm_bf16<<<cdiv(tiles, 4), 128, 0, stream>>>(xb[i], Wq_b + (size_t)i * DD,
                                                    bq + i * D_MODEL, qf,
                                                    Ni, D_MODEL, D_MODEL);
    k_gemm_bf16<<<cdiv(tiles, 4), 128, 0, stream>>>(xb[i], Wk_b + (size_t)i * DD,
                                                    bk + i * D_MODEL, kf,
                                                    Ni, D_MODEL, D_MODEL);
    // RoPE with per-scale position stride 2^i (Q pre-scaled by 1/8)
    int nrope = Ni * NHEAD * (HDIM / 2);
    k_rope_cvt<<<cdiv(nrope, 256), 256, 0, stream>>>(qf, kf, qb, kb2, Ni,
                                                     (float)(1 << i));
    // flash attention (TDM-staged V)
    dim3 ag(Ni / 16, NHEAD);
    k_flash_attn<<<ag, 32, 0, stream>>>(qb, kb2, vb[i], ctx, Ni);
    // output projection -> d_out slice
    int nctx = Ni * D_MODEL;
    k_cvt_bf16<<<cdiv(nctx, 256), 256, 0, stream>>>(ctx, ctxb, nctx);
    k_gemm_bf16<<<cdiv(tiles, 4), 128, 0, stream>>>(ctxb, Wo_b, bo, out + out_off,
                                                    Ni, D_MODEL, D_MODEL);
    out_off += (size_t)Ni * D_MODEL;
  }
}
